// GRU_75299366633930
// MI455X (gfx1250) — matrix operand order repaired and checked by reading, not compiled, not run on hardware
//
#include <hip/hip_runtime.h>
#include <stdint.h>

// ---------------------------------------------------------------------------
// GRU stack (T=64,B=64,H=E=1024,V=32000,L=2) on gfx1250 via bf16 WMMA.
// GEMM: C[M,N] = bias + A1·W1^T (+ A2·W2^T), f32 accum.
//   - block tile 64M x 256N, 8 waves, wave tile 32M x 64N (8 wmma/k-step)
//   - A slab (64x64 bf16) double-buffered in LDS via global_load_async_to_lds
//     (ASYNCcnt) + workgroup barriers; B streamed with forced global_load
//     (address_space(1)) so LOADcnt stays decoupled from DScnt.
// ---------------------------------------------------------------------------

typedef __attribute__((ext_vector_type(16))) __bf16 bf16x16;
typedef __attribute__((ext_vector_type(8)))  __bf16 bf16x8;
typedef __attribute__((ext_vector_type(8)))  float  f32x8;
typedef unsigned short u16;
typedef __attribute__((address_space(3))) unsigned short lds_u16;

#define T_DIM 64
#define B_DIM 64
#define H_DIM 1024
#define V_DIM 32000
#define L_DIM 2

#define A_PITCH 72          // 64 + 8 bf16 pad: 144B row pitch -> conflict-free
#define A_SLAB  (64 * A_PITCH)

__device__ __forceinline__ u16 f2bf_bits(float x) {
  union { float f; uint32_t u; } v; v.f = x;
  uint32_t r = v.u + 0x7FFFu + ((v.u >> 16) & 1u);   // round-to-nearest-even
  return (u16)(r >> 16);
}
__device__ __forceinline__ float sigmoidf_fast(float x) {
  return 1.0f / (1.0f + __expf(-x));
}
__device__ __forceinline__ float tanhf_fast(float x) {
  return 2.0f / (1.0f + __expf(-2.0f * x)) - 1.0f;
}
__device__ __forceinline__ f32x8 zero8() {
  f32x8 z = {0.f,0.f,0.f,0.f,0.f,0.f,0.f,0.f};
  return z;
}
__device__ __forceinline__ void wait_async0() {
#if __has_builtin(__builtin_amdgcn_s_wait_asynccnt)
  __builtin_amdgcn_s_wait_asynccnt(0);
#else
  asm volatile("s_wait_asynccnt 0x0" ::: "memory");
#endif
}
// 16B global -> LDS async copy (per-lane addresses), tracked by ASYNCcnt.
__device__ __forceinline__ void async_cp16(uint32_t lds_byte, const void* g) {
  asm volatile("global_load_async_to_lds_b128 %0, %1, off"
               :: "v"(lds_byte), "v"(g) : "memory");
}

// Forced global (addrspace 1) accesses -> global_load_b128 / global_store_b32.
__device__ __forceinline__ bf16x16 gload_bfrag(const __bf16* p) {
  typedef __attribute__((address_space(1))) const bf16x16 gv16;
  return *(gv16*)p;
}
__device__ __forceinline__ float gloadf(const float* p) {
  typedef __attribute__((address_space(1))) const float gcf;
  return *(gcf*)p;
}
__device__ __forceinline__ void gstoref(float* p, float v) {
  typedef __attribute__((address_space(1))) float gf;
  *(gf*)p = v;
}

// A-frag from LDS: lane m<16 -> K {k..k+15}; lanes 16-31 -> +16.
__device__ __forceinline__ bf16x16 make_afrag(const __bf16* p, int lh) {
  bf16x8 lo = *(const bf16x8*)(p + (lh << 4));
  bf16x8 hi = *(const bf16x8*)(p + 8 + (lh << 4));
  return __builtin_shufflevector(lo, hi,
      0,1,2,3,4,5,6,7,8,9,10,11,12,13,14,15);
}

// C[M,N] = bias[N] + A1·W1^T (+ A2·W2^T if dual).
// Grid: (N/256, M/64).  M,N,K all divide tiling exactly -> no divergence.
__global__ __launch_bounds__(256)
void wmma_gemm_dual(const u16* __restrict__ A1p, const u16* __restrict__ W1p,
                    const u16* __restrict__ A2p, const u16* __restrict__ W2p,
                    const float* __restrict__ bias, float* __restrict__ C,
                    int M, int N, int K, int dual) {
  __shared__ __align__(16) u16 sA[2][A_SLAB];

  const int tid  = threadIdx.x;
  const int lane = tid & 31;
  const int wave = tid >> 5;
  const int lr   = lane & 15;
  const int lh   = lane >> 4;                  // 0/1: K-half selector
  const int wm   = wave & 1;                   // 2 M-groups of 32 rows
  const int wn   = wave >> 1;                  // 4 N-groups of 64 cols
  const int gm   = blockIdx.y * 64;            // block row base
  const int n0   = blockIdx.x * 256 + wn * 64; // wave col base

  // LDS fill: 64 rows x 128B = 512 x 16B chunks, 2 per thread.
  const int r0  = tid >> 3,          kc0 = (tid & 7) << 3;
  const int r1  = (tid + 256) >> 3,  kc1 = ((tid + 256) & 7) << 3;
  const uint32_t ldsbase = (uint32_t)(uintptr_t)(lds_u16*)&sA[0][0];

  f32x8 acc[8];
#pragma unroll
  for (int i = 0; i < 8; ++i) acc[i] = zero8();

  auto sweep = [&](const u16* Ap, const u16* Wp) {
    const __bf16* A = (const __bf16*)Ap;
    const __bf16* W = (const __bf16*)Wp;
    const __bf16* w0 = W + (size_t)(n0 +  0 + lr) * K + (lh << 4);
    const __bf16* w1 = W + (size_t)(n0 + 16 + lr) * K + (lh << 4);
    const __bf16* w2 = W + (size_t)(n0 + 32 + lr) * K + (lh << 4);
    const __bf16* w3 = W + (size_t)(n0 + 48 + lr) * K + (lh << 4);

    auto fill = [&](int k0, int buf) {
      uint32_t l0 = ldsbase + (uint32_t)((buf * A_SLAB + r0 * A_PITCH + kc0) * 2);
      uint32_t l1 = ldsbase + (uint32_t)((buf * A_SLAB + r1 * A_PITCH + kc1) * 2);
      async_cp16(l0, (const void*)(A + (size_t)(gm + r0) * K + k0 + kc0));
      async_cp16(l1, (const void*)(A + (size_t)(gm + r1) * K + k0 + kc1));
    };

    fill(0, 0);
    wait_async0();
    __syncthreads();

    const int nslab = K >> 6;                  // 64-wide K slabs
    for (int s = 0; s < nslab; ++s) {
      const int cur = s & 1;
      if (s + 1 < nslab) fill((s + 1) << 6, cur ^ 1);

#pragma unroll
      for (int half = 0; half < 2; ++half) {
        const int kk    = half << 5;           // K offset inside slab
        const int kglob = (s << 6) + kk;       // K offset for B
        bf16x16 a0 = make_afrag(
            (const __bf16*)&sA[cur][(wm * 32 + lr) * A_PITCH + kk], lh);
        bf16x16 a1 = make_afrag(
            (const __bf16*)&sA[cur][(wm * 32 + 16 + lr) * A_PITCH + kk], lh);
        bf16x16 b0 = gload_bfrag(w0 + kglob);
        bf16x16 b1 = gload_bfrag(w1 + kglob);
        bf16x16 b2 = gload_bfrag(w2 + kglob);
        bf16x16 b3 = gload_bfrag(w3 + kglob);
        // alternate a0/a1 so both A-frags stay live (no WAR hazard NOPs)
        acc[0] = __builtin_amdgcn_wmma_f32_16x16x32_bf16(false, a0, false, b0, (short)0, acc[0], false, false);
        acc[4] = __builtin_amdgcn_wmma_f32_16x16x32_bf16(false, a1, false, b0, (short)0, acc[4], false, false);
        acc[1] = __builtin_amdgcn_wmma_f32_16x16x32_bf16(false, a0, false, b1, (short)0, acc[1], false, false);
        acc[5] = __builtin_amdgcn_wmma_f32_16x16x32_bf16(false, a1, false, b1, (short)0, acc[5], false, false);
        acc[2] = __builtin_amdgcn_wmma_f32_16x16x32_bf16(false, a0, false, b2, (short)0, acc[2], false, false);
        acc[6] = __builtin_amdgcn_wmma_f32_16x16x32_bf16(false, a1, false, b2, (short)0, acc[6], false, false);
        acc[3] = __builtin_amdgcn_wmma_f32_16x16x32_bf16(false, a0, false, b3, (short)0, acc[3], false, false);
        acc[7] = __builtin_amdgcn_wmma_f32_16x16x32_bf16(false, a1, false, b3, (short)0, acc[7], false, false);
      }
      if (s + 1 < nslab) { wait_async0(); __syncthreads(); }
    }
    __syncthreads();                           // protect LDS reuse across passes
  };

  sweep(A1p, W1p);
  if (dual) sweep(A2p, W2p);

  // Epilogue: D rows = base + (lh?8:0) + r, cols = n-tile + lr.
#pragma unroll
  for (int f = 0; f < 2; ++f) {
    const int mrow = gm + wm * 32 + f * 16 + (lh << 3);
#pragma unroll
    for (int t = 0; t < 4; ++t) {
      const float bv = gloadf(bias + n0 + t * 16 + lr);
      const f32x8 a = acc[f * 4 + t];
#pragma unroll
      for (int r = 0; r < 8; ++r) {
        gstoref(C + (size_t)(mrow + r) * N + n0 + t * 16 + lr, a[r] + bv);
      }
    }
  }
}

// ------------------------------ aux kernels --------------------------------
__global__ void cvt_bf16_k(const float* __restrict__ s, u16* __restrict__ d, size_t n) {
  size_t i = (size_t)blockIdx.x * blockDim.x + threadIdx.x;
  if (i < n) d[i] = f2bf_bits(s[i]);
}
__global__ void addbias_k(const float* __restrict__ a, const float* __restrict__ b,
                          float* __restrict__ d, int n) {
  int i = blockIdx.x * blockDim.x + threadIdx.x;
  if (i < n) d[i] = a[i] + b[i];
}
__global__ void gather_emb_k(const int* __restrict__ tok, const float* __restrict__ emb,
                             u16* __restrict__ xe, size_t n) {
  size_t i = (size_t)blockIdx.x * blockDim.x + threadIdx.x;
  if (i < n) {
    size_t tb = i >> 10;          // H = 1024
    int e = (int)(i & 1023);
    int tk = tok[tb];
    xe[i] = f2bf_bits(emb[(size_t)tk * H_DIM + e]);
  }
}
__global__ void init_h_k(const float* __restrict__ hid, float* __restrict__ h,
                         u16* __restrict__ hbf, int n) {
  int i = blockIdx.x * blockDim.x + threadIdx.x;
  if (i < n) { float v = hid[i]; h[i] = v; hbf[i] = f2bf_bits(v); }
}
__global__ void rep_x_k(const u16* __restrict__ src, u16* __restrict__ dst, int n) {
  int i = blockIdx.x * blockDim.x + threadIdx.x;
  if (i < n) dst[i] = src[i & (B_DIM * H_DIM - 1)];
}
__global__ void ew_rf_k(const float* __restrict__ pre, const float* __restrict__ h,
                        float* __restrict__ f, u16* __restrict__ rh, int n) {
  int i = blockIdx.x * blockDim.x + threadIdx.x;
  if (i < n) {
    int row = i >> 10;
    int col = i & 1023;
    const float* pr = pre + ((size_t)row << 11);
    float rv = sigmoidf_fast(pr[col]);
    float fv = sigmoidf_fast(pr[col + H_DIM]);
    f[i]  = fv;
    rh[i] = f2bf_bits(rv * h[i]);
  }
}
__global__ void ew_h_k(const float* __restrict__ pre, const float* __restrict__ f,
                       float* __restrict__ h, u16* __restrict__ hbf, int n) {
  int i = blockIdx.x * blockDim.x + threadIdx.x;
  if (i < n) {
    float hw = tanhf_fast(pre[i]);
    float hv = h[i];
    float hn = hv + f[i] * (hw - hv);
    h[i] = hn;
    hbf[i] = f2bf_bits(hn);
  }
}
__global__ void copy_f32_k(const float* __restrict__ s, float* __restrict__ d, int n) {
  int i = blockIdx.x * blockDim.x + threadIdx.x;
  if (i < n) d[i] = s[i];
}

// ------------------------------ host driver --------------------------------
extern "C" void kernel_launch(void* const* d_in, const int* in_sizes, int n_in,
                              void* d_out, int out_size, void* d_ws, size_t ws_size,
                              hipStream_t stream) {
  (void)in_sizes; (void)n_in; (void)out_size; (void)ws_size;
  const int*   tokens = (const int*)  d_in[0];
  const float* hidden = (const float*)d_in[1];
  const float* emb    = (const float*)d_in[2];
  const float* Wr  = (const float*)d_in[3];  const float* br   = (const float*)d_in[4];
  const float* Wf  = (const float*)d_in[5];  const float* bf_  = (const float*)d_in[6];
  const float* Wh  = (const float*)d_in[7];  const float* bh   = (const float*)d_in[8];
  const float* Ur  = (const float*)d_in[9];  const float* bur  = (const float*)d_in[10];
  const float* Uf  = (const float*)d_in[11]; const float* buf_ = (const float*)d_in[12];
  const float* Uh  = (const float*)d_in[13]; const float* buh  = (const float*)d_in[14];
  const float* Wdec = (const float*)d_in[15]; const float* bdec = (const float*)d_in[16];
  float* out = (float*)d_out;

  const size_t HH = (size_t)H_DIM * H_DIM;
  char* p = (char*)d_ws;
  auto carve = [&](size_t bytes) -> char* {
    char* r = p; p += (bytes + 255) & ~(size_t)255; return r;
  };
  u16*   Wrf    = (u16*)  carve((size_t)L_DIM * 2 * HH * 2);
  u16*   Urf    = (u16*)  carve((size_t)L_DIM * 2 * HH * 2);
  u16*   Whb    = (u16*)  carve((size_t)L_DIM * HH * 2);
  u16*   Uhb    = (u16*)  carve((size_t)L_DIM * HH * 2);
  u16*   Wdb    = (u16*)  carve((size_t)V_DIM * H_DIM * 2);
  float* biasRF = (float*)carve((size_t)L_DIM * 2 * H_DIM * 4);
  float* biasH  = (float*)carve((size_t)L_DIM * H_DIM * 4);
  u16*   xe     = (u16*)  carve((size_t)T_DIM * B_DIM * H_DIM * 2);
  u16*   x0     = (u16*)  carve((size_t)L_DIM * B_DIM * H_DIM * 2);
  float* hst    = (float*)carve((size_t)L_DIM * B_DIM * H_DIM * 4);
  u16*   hbf    = (u16*)  carve((size_t)L_DIM * B_DIM * H_DIM * 2);
  float* preRF  = (float*)carve((size_t)L_DIM * B_DIM * 2 * H_DIM * 4);
  float* preH   = (float*)carve((size_t)L_DIM * B_DIM * H_DIM * 4);
  float* fbuf   = (float*)carve((size_t)L_DIM * B_DIM * H_DIM * 4);
  u16*   rh     = (u16*)  carve((size_t)L_DIM * B_DIM * H_DIM * 2);

  auto cvt = [&](const float* s, u16* d, size_t n) {
    cvt_bf16_k<<<dim3((unsigned)((n + 255) / 256)), 256, 0, stream>>>(s, d, n);
  };
  auto addb = [&](const float* a, const float* b, float* d, int n) {
    addbias_k<<<dim3((n + 255) / 256), 256, 0, stream>>>(a, b, d, n);
  };

  for (int l = 0; l < L_DIM; ++l) {
    cvt(Wr + l * HH, Wrf + (size_t)l * 2 * HH,      HH);
    cvt(Wf + l * HH, Wrf + (size_t)l * 2 * HH + HH, HH);
    cvt(Ur + l * HH, Urf + (size_t)l * 2 * HH,      HH);
    cvt(Uf + l * HH, Urf + (size_t)l * 2 * HH + HH, HH);
    cvt(Wh + l * HH, Whb + (size_t)l * HH, HH);
    cvt(Uh + l * HH, Uhb + (size_t)l * HH, HH);
    addb(br  + l * H_DIM, bur  + l * H_DIM, biasRF + (size_t)l * 2 * H_DIM,         H_DIM);
    addb(bf_ + l * H_DIM, buf_ + l * H_DIM, biasRF + (size_t)l * 2 * H_DIM + H_DIM, H_DIM);
    addb(bh  + l * H_DIM, buh  + l * H_DIM, biasH  + (size_t)l * H_DIM,             H_DIM);
  }
  cvt(Wdec, Wdb, (size_t)V_DIM * H_DIM);
  {
    size_t n = (size_t)T_DIM * B_DIM * H_DIM;
    gather_emb_k<<<dim3((unsigned)((n + 255) / 256)), 256, 0, stream>>>(tokens, emb, xe, n);
  }
  {
    int n = L_DIM * B_DIM * H_DIM;
    init_h_k<<<dim3((n + 255) / 256), 256, 0, stream>>>(hidden, hst, hbf, n);
  }

  auto gemm = [&](const u16* A1, const u16* W1, const u16* A2, const u16* W2,
                  const float* bias, float* C, int M, int N, int dual) {
    dim3 g(N / 256, M / 64);
    wmma_gemm_dual<<<g, 256, 0, stream>>>(A1, W1, A2, W2, bias, C, M, N, H_DIM, dual);
  };

  const int rows = L_DIM * B_DIM;                 // 128
  const int nEl  = rows * H_DIM;                  // 131072
  const dim3 gEl((nEl + 255) / 256);

  for (int t = 0; t < T_DIM; ++t) {
    rep_x_k<<<gEl, 256, 0, stream>>>(xe + (size_t)t * B_DIM * H_DIM, x0, nEl);
    for (int l = 0; l < L_DIM; ++l) {
      const u16* xin = (l == 0) ? x0 : hbf;
      gemm(xin, Wrf + (size_t)l * 2 * HH, hbf, Urf + (size_t)l * 2 * HH,
           biasRF + (size_t)l * 2 * H_DIM, preRF, rows, 2 * H_DIM, 1);
      ew_rf_k<<<gEl, 256, 0, stream>>>(preRF, hst, fbuf, rh, nEl);
      gemm(xin, Whb + (size_t)l * HH, rh, Uhb + (size_t)l * HH,
           biasH + (size_t)l * H_DIM, preH, rows, H_DIM, 1);
      ew_h_k<<<gEl, 256, 0, stream>>>(preH, fbuf, hst, hbf, nEl);
    }
    gemm(hbf + (size_t)(L_DIM - 1) * B_DIM * H_DIM, Wdb,
         hbf, Wdb, bdec, out + (size_t)t * B_DIM * V_DIM, B_DIM, V_DIM, 0);
  }
  copy_f32_k<<<gEl, 256, 0, stream>>>(hst, out + (size_t)T_DIM * B_DIM * V_DIM, nEl);
}
